// MultiHeadAttention_31636729102987
// MI455X (gfx1250) — matrix operand order repaired and checked by reading, not compiled, not run on hardware
//
#include <hip/hip_runtime.h>
#include <hip/hip_bf16.h>

typedef _Float16 h16;
typedef __attribute__((ext_vector_type(4)))  _Float16 v4h;
typedef __attribute__((ext_vector_type(16))) _Float16 v16h;
typedef __attribute__((ext_vector_type(8)))  float    v8f;
typedef __attribute__((ext_vector_type(4)))  int      i4;
typedef __attribute__((address_space(1)))    i4       gi4;   // global int4
typedef __attribute__((address_space(3)))    i4       li4;   // LDS int4

#define EMBED 1024
#define SEQ   2048
#define BATCH 8
#define MTOT  (BATCH*SEQ)

// ---- optional CDNA5 async global->LDS path (guarded; falls back cleanly) ----
#if defined(__has_builtin)
#  if __has_builtin(__builtin_amdgcn_global_load_async_to_lds_b128)
#    define HAVE_ASYNC_LDS 1
#  endif
#endif
#ifndef HAVE_ASYNC_LDS
#  define HAVE_ASYNC_LDS 0
#endif

__device__ __forceinline__ void wait_async0() {
#if defined(__has_builtin) && __has_builtin(__builtin_amdgcn_s_wait_asynccnt)
    __builtin_amdgcn_s_wait_asynccnt(0);
#else
    asm volatile("s_wait_asynccnt 0x0" ::: "memory");
#endif
}

// Assemble a v16h WMMA operand from two 16B-aligned 16-byte pieces.
__device__ __forceinline__ v16h frag2(const h16* p0, const h16* p1) {
    v16h r;
    *(float4*)(&r)       = *(const float4*)p0;
    *((float4*)(&r) + 1) = *(const float4*)p1;
    return r;
}
// Assemble from one contiguous 32-byte piece.
__device__ __forceinline__ v16h frag1(const h16* p) {
    v16h r;
    *(float4*)(&r)       = *(const float4*)p;
    *((float4*)(&r) + 1) = *((const float4*)p + 1);
    return r;
}

// ---------------- conversion kernels ----------------
__global__ __launch_bounds__(256) void cvt_f32_f16(const float* __restrict__ in,
                                                   h16* __restrict__ out, int n4) {
    int i = blockIdx.x * 256 + threadIdx.x;
    if (i < n4) {
        float4 v = ((const float4*)in)[i];
        v4h r;
        r[0] = (h16)v.x; r[1] = (h16)v.y; r[2] = (h16)v.z; r[3] = (h16)v.w;
        *(v4h*)(out + 4 * (size_t)i) = r;
    }
}

// Wt[n][k] = (f16) W[k][n]   (1024x1024)
__global__ __launch_bounds__(256) void cvt_w_t(const float* __restrict__ W,
                                               h16* __restrict__ Wt) {
    int idx = blockIdx.x * 256 + threadIdx.x;   // 1M elements
    int n = idx >> 10, k = idx & 1023;
    Wt[idx] = (h16)W[(k << 10) + n];
}

// ---------------- WMMA GEMM: C = (A @ Bt^T + bias) * cscale ----------------
// A: [M,1024] f16 row-major.  Bt: [N=1024,1024] f16 (B transposed, K-contig).
// MODE 0: f16 row-major out. MODE 1: f16 out transposed to [b][e][s]. MODE 2: f32 out.
template<int MODE>
__global__ __launch_bounds__(256) void gemm_wmma(
    const h16* __restrict__ A, const h16* __restrict__ Bt,
    const float* __restrict__ bias, float cscale,
    h16* __restrict__ Oh, float* __restrict__ Of)
{
    __shared__ __align__(16) h16 As[2][128][40];   // double-buffered, rows 16B aligned
    __shared__ __align__(16) h16 Bs[2][128][40];

    const int tid  = threadIdx.x;
    const int lane = tid & 31, w = tid >> 5;
    const int wm = w & 3, wn = w >> 2;            // 4(M) x 2(N) wave grid
    const int m0 = blockIdx.y * 128, n0 = blockIdx.x * 128;
    const int lrow = lane & 15, lhi = lane >> 4;
    const int koff = lhi * 8;                     // unused (A-frag also uses kb)
    const int kb   = lhi * 16;                    // B-frag K offset
    const int row0 = tid >> 2, seg0 = (tid & 3) * 8;  // this thread's 16B chunk

    v8f acc[2][4];
    #pragma unroll
    for (int i = 0; i < 2; i++)
        #pragma unroll
        for (int j = 0; j < 4; j++) acc[i][j] = (v8f){};

#if HAVE_ASYNC_LDS
    auto stage = [&](int k0, int buf) {
        #pragma unroll
        for (int c = 0; c < 2; c++) {
            int row = row0 + 64 * c;
            __builtin_amdgcn_global_load_async_to_lds_b128(
                (gi4*)(void*)&A [(size_t)(m0 + row) * EMBED + k0 + seg0],
                (li4*)&As[buf][row][seg0], 0, 0);
            __builtin_amdgcn_global_load_async_to_lds_b128(
                (gi4*)(void*)&Bt[(size_t)(n0 + row) * EMBED + k0 + seg0],
                (li4*)&Bs[buf][row][seg0], 0, 0);
        }
    };
    stage(0, 0);
#else
    float4 ra[2], rb[2];
    auto stage_load = [&](int k0) {
        #pragma unroll
        for (int c = 0; c < 2; c++) {
            int row = row0 + 64 * c;
            ra[c] = *(const float4*)&A [(size_t)(m0 + row) * EMBED + k0 + seg0];
            rb[c] = *(const float4*)&Bt[(size_t)(n0 + row) * EMBED + k0 + seg0];
        }
    };
    auto stage_store = [&](int buf) {
        #pragma unroll
        for (int c = 0; c < 2; c++) {
            int row = row0 + 64 * c;
            *(float4*)&As[buf][row][seg0] = ra[c];
            *(float4*)&Bs[buf][row][seg0] = rb[c];
        }
    };
    stage_load(0); stage_store(0);
#endif

    for (int kk = 0; kk < 32; kk++) {
        const int buf = kk & 1;
#if HAVE_ASYNC_LDS
        wait_async0();           // asyncs filling `buf` (issued last iter) done
        __syncthreads();
        if (kk + 1 < 32) stage((kk + 1) * 32, buf ^ 1);   // overlaps compute
#else
        __syncthreads();
        if (kk + 1 < 32) stage_load((kk + 1) * 32);       // global loads overlap compute
#endif
        v16h af[2], bf[4];
        #pragma unroll
        for (int i = 0; i < 2; i++) {
            const h16* p = &As[buf][wm * 32 + i * 16 + lrow][0];
            af[i] = frag1(p + kb);
        }
        #pragma unroll
        for (int j = 0; j < 4; j++)
            bf[j] = frag1(&Bs[buf][wn * 64 + j * 16 + lrow][kb]);

        #pragma unroll
        for (int i = 0; i < 2; i++)
            #pragma unroll
            for (int j = 0; j < 4; j++)
                acc[i][j] = __builtin_amdgcn_wmma_f32_16x16x32_f16(
                    false, af[i], false, bf[j], (short)0, acc[i][j], false, false);
#if !HAVE_ASYNC_LDS
        if (kk + 1 < 32) stage_store(buf ^ 1);
#endif
    }

    #pragma unroll
    for (int j = 0; j < 4; j++) {
        int col = n0 + wn * 64 + j * 16 + lrow;
        float bv = bias ? bias[col] : 0.f;
        #pragma unroll
        for (int i = 0; i < 2; i++) {
            int rbase = m0 + wm * 32 + i * 16 + lhi * 8;
            #pragma unroll
            for (int r = 0; r < 8; r++) {
                float v = (acc[i][j][r] + bv) * cscale;
                int row = rbase + r;
                if (MODE == 0)
                    Oh[(size_t)row * EMBED + col] = (h16)v;
                else if (MODE == 1)  // [b][e][s]
                    Oh[((size_t)((row >> 11) * EMBED + col)) * SEQ + (row & (SEQ - 1))] = (h16)v;
                else
                    Of[(size_t)row * EMBED + col] = v;
            }
        }
    }
}

// ---------------- flash attention ----------------
// Q,K f16 [b][s][e] (Q pre-scaled by 1/32), Vt f16 [b][e][s], O f16 [b][s][e].
// Block = 32 q rows: 8 waves = 2 row-groups(16) x 4 col-groups(256 of E).
__global__ __launch_bounds__(256) void attn_wmma(
    const h16* __restrict__ Q, const h16* __restrict__ K,
    const h16* __restrict__ Vt, h16* __restrict__ O)
{
    __shared__ __align__(16) float Ssh[2][16][64];   // split-K reduced scores
    __shared__ __align__(16) h16   Psh[2][16][72];   // probs, padded rows (144B)
    __shared__ float m_s[2][16], l_s[2][16], a_s[2][16];

    const int tid = threadIdx.x, lane = tid & 31, w = tid >> 5;
    const int rg = w >> 2, cg = w & 3;
    const int b  = blockIdx.x >> 6;
    const int s0 = (blockIdx.x & 63) * 32;
    const int lrow = lane & 15, lhi = lane >> 4;
    const int koff = lhi * 8, kb = lhi * 16;

    const h16* Qb = Q  + (size_t)(b * SEQ + s0 + rg * 16) * EMBED;
    const h16* Kb = K  + (size_t)b * SEQ * EMBED;
    const h16* Vb = Vt + (size_t)b * EMBED * SEQ;

    if (tid < 32) { m_s[tid >> 4][tid & 15] = -1e30f; l_s[tid >> 4][tid & 15] = 0.f; }

    v8f oacc[16];
    #pragma unroll
    for (int j = 0; j < 16; j++) oacc[j] = (v8f){};

    auto ldQ = [&](int ke) -> v16h {
        const h16* qp = Qb + (size_t)lrow * EMBED + cg * 256 + ke * 32;
        return frag1(qp + kb);
    };
    auto ldK = [&](int t0, int ke, int j) -> v16h {
        return frag1(Kb + (size_t)(t0 + j * 16 + lrow) * EMBED + cg * 256 + ke * 32 + kb);
    };
    auto ldP = [&](int kk) -> v16h {
        const h16* pp = &Psh[rg][lrow][kk * 32];
        return frag1(pp + kb);
    };
    auto ldV = [&](int t0, int kk, int j) -> v16h {
        return frag1(Vb + (size_t)(cg * 256 + j * 16 + lrow) * SEQ + t0 + kk * 32 + kb);
    };

    for (int kc = 0; kc < SEQ / 64; kc++) {
        const int t0 = kc * 64;

        // hint-prefetch next chunk's K/V streams
        if (kc + 1 < SEQ / 64) {
            const int tn = t0 + 64;
            const h16* kp = Kb + (size_t)(tn + lane * 2) * EMBED + cg * 256;
            __builtin_prefetch(kp, 0, 0);
            __builtin_prefetch(kp + EMBED, 0, 0);
            #pragma unroll
            for (int pv = 0; pv < 8; pv++)
                __builtin_prefetch(Vb + (size_t)(cg * 256 + pv * 32 + lane) * SEQ + tn, 0, 0);
        }

        for (int z = tid; z < 2 * 16 * 64; z += 256) ((float*)Ssh)[z] = 0.f;
        __syncthreads();

        // ---- partial scores over this wave's 256-wide E slice (pipelined) ----
        v8f sacc[4];
        #pragma unroll
        for (int j = 0; j < 4; j++) sacc[j] = (v8f){};
        {
            v16h aq = ldQ(0);
            v16h bk = ldK(t0, 0, 0);
            #pragma unroll
            for (int ke = 0; ke < 8; ke++) {
                v16h aq_n = aq;
                if (ke < 7) aq_n = ldQ(ke + 1);
                #pragma unroll
                for (int j = 0; j < 4; j++) {
                    v16h bk_n = bk;
                    if (!(ke == 7 && j == 3))
                        bk_n = (j < 3) ? ldK(t0, ke, j + 1) : ldK(t0, ke + 1, 0);
                    sacc[j] = __builtin_amdgcn_wmma_f32_16x16x32_f16(
                        false, aq, false, bk, (short)0, sacc[j], false, false);
                    bk = bk_n;
                }
                aq = aq_n;
            }
        }
        #pragma unroll
        for (int j = 0; j < 4; j++)
            #pragma unroll
            for (int r = 0; r < 8; r++)
                atomicAdd(&Ssh[rg][r + lhi * 8][j * 16 + lrow], sacc[j][r]);
        __syncthreads();

        // ---- online softmax: one wave per row group, lane pair per row ----
        if (cg == 0) {
            int row = lane >> 1, c0 = (lane & 1) * 32;
            float vals[32];
            #pragma unroll
            for (int q8 = 0; q8 < 8; q8++) {
                float4 f = *(float4*)&Ssh[rg][row][c0 + q8 * 4];
                vals[q8*4+0] = f.x; vals[q8*4+1] = f.y;
                vals[q8*4+2] = f.z; vals[q8*4+3] = f.w;
            }
            float mx = -1e30f;
            #pragma unroll
            for (int i2 = 0; i2 < 32; i2++) mx = fmaxf(mx, vals[i2]);
            mx = fmaxf(mx, __shfl_xor(mx, 1));
            float mo = m_s[rg][row];
            float mn = fmaxf(mo, mx);
            float al = __expf(mo - mn);
            float sum = 0.f;
            #pragma unroll
            for (int i2 = 0; i2 < 32; i2++) { float p = __expf(vals[i2] - mn); vals[i2] = p; sum += p; }
            sum += __shfl_xor(sum, 1);
            if ((lane & 1) == 0) {
                m_s[rg][row] = mn;
                l_s[rg][row] = l_s[rg][row] * al + sum;
                a_s[rg][row] = al;
            }
            #pragma unroll
            for (int i2 = 0; i2 < 32; i2++) Psh[rg][row][c0 + i2] = (h16)vals[i2];
        }
        __syncthreads();

        // ---- rescale running output, then O += P @ V (pipelined) ----
        float a8[8];
        #pragma unroll
        for (int r = 0; r < 8; r++) a8[r] = a_s[rg][lhi * 8 + r];
        #pragma unroll
        for (int j = 0; j < 16; j++)
            #pragma unroll
            for (int r = 0; r < 8; r++) oacc[j][r] *= a8[r];

        {
            v16h ap = ldP(0);
            v16h bv = ldV(t0, 0, 0);
            #pragma unroll
            for (int kk = 0; kk < 2; kk++) {
                v16h ap_n = ap;
                if (kk == 0) ap_n = ldP(1);
                #pragma unroll
                for (int j = 0; j < 16; j++) {
                    v16h bv_n = bv;
                    if (!(kk == 1 && j == 15))
                        bv_n = (j < 15) ? ldV(t0, kk, j + 1) : ldV(t0, kk + 1, 0);
                    oacc[j] = __builtin_amdgcn_wmma_f32_16x16x32_f16(
                        false, ap, false, bv, (short)0, oacc[j], false, false);
                    bv = bv_n;
                }
                ap = ap_n;
            }
        }
        __syncthreads();
    }

    // finalize: divide by l, store f16 [b][s][e]
    float li[8];
    #pragma unroll
    for (int r = 0; r < 8; r++) li[r] = 1.f / l_s[rg][lhi * 8 + r];
    h16* Ob = O + (size_t)(b * SEQ + s0 + rg * 16) * EMBED;
    #pragma unroll
    for (int j = 0; j < 16; j++) {
        int col = cg * 256 + j * 16 + lrow;
        #pragma unroll
        for (int r = 0; r < 8; r++)
            Ob[(size_t)(lhi * 8 + r) * EMBED + col] = (h16)(oacc[j][r] * li[r]);
    }
}

// ---------------- launcher ----------------
extern "C" void kernel_launch(void* const* d_in, const int* in_sizes, int n_in,
                              void* d_out, int out_size, void* d_ws, size_t ws_size,
                              hipStream_t stream) {
    const float* x  = (const float*)d_in[0];
    const float* Wq = (const float*)d_in[1];
    const float* bq = (const float*)d_in[2];
    const float* Wk = (const float*)d_in[3];
    const float* bk = (const float*)d_in[4];
    const float* Wv = (const float*)d_in[5];
    const float* bv = (const float*)d_in[6];
    const float* Wo = (const float*)d_in[7];
    const float* bo = (const float*)d_in[8];
    float* out = (float*)d_out;

    char* ws = (char*)d_ws;
    h16* x16 = (h16*)(ws);                          // 32 MB
    h16* wqT = (h16*)(ws + ((size_t)32 << 20));     // 2 MB each
    h16* wkT = (h16*)(ws + ((size_t)34 << 20));
    h16* wvT = (h16*)(ws + ((size_t)36 << 20));
    h16* woT = (h16*)(ws + ((size_t)38 << 20));
    h16* q16 = (h16*)(ws + ((size_t)40 << 20));     // 32 MB
    h16* k16 = (h16*)(ws + ((size_t)72 << 20));     // 32 MB
    h16* vT  = (h16*)(ws + ((size_t)104 << 20));    // 32 MB ([b][e][s])
    h16* a16 = (h16*)(ws + ((size_t)136 << 20));    // 32 MB

    const int n  = MTOT * EMBED;                    // 16.7M
    cvt_f32_f16<<<(n / 4) / 256, 256, 0, stream>>>(x, x16, n / 4);
    const int nw = (EMBED * EMBED) / 256;
    cvt_w_t<<<nw, 256, 0, stream>>>(Wq, wqT);
    cvt_w_t<<<nw, 256, 0, stream>>>(Wk, wkT);
    cvt_w_t<<<nw, 256, 0, stream>>>(Wv, wvT);
    cvt_w_t<<<nw, 256, 0, stream>>>(Wo, woT);

    dim3 gg(EMBED / 128, MTOT / 128);
    gemm_wmma<0><<<gg, 256, 0, stream>>>(x16, wqT, bq, 0.03125f, q16, nullptr); // q, pre-scaled 1/sqrt(E)
    gemm_wmma<0><<<gg, 256, 0, stream>>>(x16, wkT, bk, 1.0f,     k16, nullptr);
    gemm_wmma<1><<<gg, 256, 0, stream>>>(x16, wvT, bv, 1.0f,     vT,  nullptr); // v transposed

    attn_wmma<<<MTOT / 32, 256, 0, stream>>>(q16, k16, vT, a16);

    gemm_wmma<2><<<gg, 256, 0, stream>>>(a16, woT, bo, 1.0f, nullptr, out);     // + bo, f32 out
}
